// DQN_38448547234332
// MI455X (gfx1250) — matrix operand order repaired and checked by reading, not compiled, not run on hardware
//
#include <hip/hip_runtime.h>
#include <hip/hip_bf16.h>

// Problem constants (fixed by the reference)
#define IN_DIM  128
#define HID     24
#define OUT_DIM 4096
#define BATCH   16384
#define KMOVES  256

typedef __attribute__((ext_vector_type(4)))  float     f32x4;
typedef __attribute__((ext_vector_type(8)))  float     v8f;
typedef __attribute__((ext_vector_type(16))) _Float16  v16h;

// ---------------------------------------------------------------------------
// Kernel 1: per-row MLP layers 1+2 (f32, exact), store h as f16 [row][32]
// (K padded 24->32 with zeros). Row layout is exactly what the WMMA A-frag
// loads want (two b128 per lane).
// ---------------------------------------------------------------------------
__global__ __launch_bounds__(256) void mlp_hidden_kernel(
    const float* __restrict__ x, const float* __restrict__ W1,
    const float* __restrict__ b1, const float* __restrict__ W2,
    const float* __restrict__ b2, _Float16* __restrict__ hbuf)
{
    __shared__ float sW1[IN_DIM * HID];
    __shared__ float sW2[HID * HID];
    __shared__ float sb1[HID];
    __shared__ float sb2[HID];

    const int tid = threadIdx.x;
    for (int i = tid; i < IN_DIM * HID; i += 256) sW1[i] = W1[i];
    for (int i = tid; i < HID * HID;  i += 256) sW2[i] = W2[i];
    if (tid < HID) { sb1[tid] = b1[tid]; sb2[tid] = b2[tid]; }
    __syncthreads();

    const int row = blockIdx.x * 256 + tid;
    const float* xr = x + (size_t)row * IN_DIM;

    float acc[HID];
#pragma unroll
    for (int j = 0; j < HID; ++j) acc[j] = sb1[j];
    for (int k = 0; k < IN_DIM; ++k) {
        const float xv = xr[k];
#pragma unroll
        for (int j = 0; j < HID; ++j) acc[j] = fmaf(xv, sW1[k * HID + j], acc[j]);
    }
#pragma unroll
    for (int j = 0; j < HID; ++j) acc[j] = fmaxf(acc[j], 0.f);   // relu

    float acc2[HID];
#pragma unroll
    for (int j = 0; j < HID; ++j) acc2[j] = sb2[j];
#pragma unroll
    for (int i = 0; i < HID; ++i) {
        const float hv = acc[i];
#pragma unroll
        for (int j = 0; j < HID; ++j) acc2[j] = fmaf(hv, sW2[i * HID + j], acc2[j]);
    }

    union { _Float16 h[32]; f32x4 f[4]; } u;
#pragma unroll
    for (int j = 0; j < HID; ++j) u.h[j] = (_Float16)fmaxf(acc2[j], 0.f);
#pragma unroll
    for (int j = HID; j < 32; ++j) u.h[j] = (_Float16)0.f;

    f32x4* dst = (f32x4*)(hbuf + (size_t)row * 32);
#pragma unroll
    for (int q = 0; q < 4; ++q) dst[q] = u.f[q];
}

// ---------------------------------------------------------------------------
// Kernel 2: repack W3 (f32 [24][4096]) into f16 WMMA B-fragments, K padded
// to 32. Layout: w3p[tile t][lane l][16 f16], where lane l<16 holds column
// n0+l for K=0..15 and lane l>=16 holds column n0+(l-16) for K=16..31
// (2 K-values per dword, low half = even position).
// ---------------------------------------------------------------------------
__global__ __launch_bounds__(256) void prep_w3_kernel(
    const float* __restrict__ W3, _Float16* __restrict__ w3p)
{
    const int tid = blockIdx.x * 256 + threadIdx.x;   // 8192 total
    const int t = tid >> 5;
    const int l = tid & 31;
    const int col   = t * 16 + (l & 15);
    const int kbase = (l >> 4) << 4;                  // 0 or 16

    union { _Float16 h[16]; f32x4 f[2]; } u;
#pragma unroll
    for (int p = 0; p < 16; ++p) {
        const int K = kbase + p;
        u.h[p] = (K < HID) ? (_Float16)W3[(size_t)K * OUT_DIM + col] : (_Float16)0.f;
    }
    f32x4* dst = (f32x4*)(w3p + (size_t)tid * 16);
    dst[0] = u.f[0];
    dst[1] = u.f[1];
}

// ---------------------------------------------------------------------------
// Kernel 3: fused GEMM (WMMA) + legal-move mask + full-dim softmax.
// Block = 256 threads (8 waves) owns 16 rows; each wave handles 32 of the
// 256 N-tiles. Two WMMA passes: (1) online (max,sum) stats, (2) write
// normalized exp. Logits never touch HBM.
// ---------------------------------------------------------------------------
__global__ __launch_bounds__(256) void gemm_mask_softmax_kernel(
    const _Float16* __restrict__ hbuf, const _Float16* __restrict__ w3p,
    const float* __restrict__ b3, const int* __restrict__ moves,
    float* __restrict__ out)
{
    __shared__ unsigned int maskBits[16 * 128];   // 16 rows x 4096 bits
    __shared__ float redM[16][8];
    __shared__ float redS[16][8];
    __shared__ float finM[16];
    __shared__ float finIS[16];

    const int tid     = threadIdx.x;
    const int wave    = tid >> 5;
    const int lane    = tid & 31;
    const int hi      = lane >> 4;      // 0: rows 0..7, 1: rows 8..15
    const int nlane   = lane & 15;      // output column within tile
    const int rowBase = blockIdx.x * 16;

    // -- build legal-move bitmask in LDS --
    for (int i = tid; i < 16 * 128; i += 256) maskBits[i] = 0u;
    __syncthreads();
    for (int i = tid; i < 16 * KMOVES; i += 256) {
        const int r   = i >> 8;
        const int col = moves[(size_t)(rowBase + r) * KMOVES + (i & (KMOVES - 1))];
        atomicOr(&maskBits[r * 128 + (col >> 5)], 1u << (col & 31));
    }

    // -- load A fragment (same 16 rows for every wave in the block) --
    const f32x4* hrow = (const f32x4*)(hbuf + (size_t)(rowBase + nlane) * 32);
    v16h afrag;
    {
        union { v16h h; f32x4 f[2]; } ua;
        ua.f[0] = hrow[2 * hi];    // K 0..7   (lanes<16) / K 16..23 (lanes>=16)
        ua.f[1] = hrow[2 * hi + 1]; // K 8..15  (lanes<16) / K 24..31 (lanes>=16)
        afrag = ua.h;
    }
    __syncthreads();

    // -- pass 1: online (m, s) per row slice --
    float m[8], s[8];
#pragma unroll
    for (int r = 0; r < 8; ++r) { m[r] = 0.f; s[r] = 0.f; }  // max >= 0 always valid offset

    for (int t = wave; t < OUT_DIM / 16; t += 8) {
        v16h bfrag;
        {
            union { v16h h; f32x4 f[2]; } ub;
            const f32x4* bp = (const f32x4*)(w3p + ((size_t)t * 32 + lane) * 16);
            ub.f[0] = bp[0];
            ub.f[1] = bp[1];
            bfrag = ub.h;
        }
        const int   col  = t * 16 + nlane;
        const float bias = b3[col];
        v8f c;
#pragma unroll
        for (int r = 0; r < 8; ++r) c[r] = bias;   // fold b3 into accumulator

        c = __builtin_amdgcn_wmma_f32_16x16x32_f16(
                false, afrag, false, bfrag, (short)0, c, false, false);

        const int colw = col >> 5;
        const int colb = col & 31;
#pragma unroll
        for (int r = 0; r < 8; ++r) {
            const int row = hi * 8 + r;
            const unsigned int bit = (maskBits[row * 128 + colw] >> colb) & 1u;
            const float f  = bit ? c[r] : 0.f;     // multiply-mask semantics
            const float d  = f - m[r];
            const float e  = __expf(fminf(d, -d)); // exp(-|d|), one exp/elem
            if (d > 0.f) { s[r] = fmaf(s[r], e, 1.f); m[r] = f; }
            else         { s[r] = s[r] + e; }
        }
    }

    // -- reduce (m,s) across the 16 lanes sharing each row --
#pragma unroll
    for (int k = 1; k < 16; k <<= 1) {
#pragma unroll
        for (int r = 0; r < 8; ++r) {
            const float mo = __shfl_xor(m[r], k, 32);
            const float so = __shfl_xor(s[r], k, 32);
            const float mn = fmaxf(m[r], mo);
            s[r] = s[r] * __expf(m[r] - mn) + so * __expf(mo - mn);
            m[r] = mn;
        }
    }
    if (nlane == 0) {
#pragma unroll
        for (int r = 0; r < 8; ++r) {
            redM[hi * 8 + r][wave] = m[r];
            redS[hi * 8 + r][wave] = s[r];
        }
    }
    __syncthreads();

    // -- reduce across the 8 waves --
    if (tid < 16) {
        float M = redM[tid][0], S = redS[tid][0];
#pragma unroll
        for (int w = 1; w < 8; ++w) {
            const float mo = redM[tid][w];
            const float so = redS[tid][w];
            const float mn = fmaxf(M, mo);
            S = S * __expf(M - mn) + so * __expf(mo - mn);
            M = mn;
        }
        finM[tid]  = M;
        finIS[tid] = 1.f / S;
    }
    __syncthreads();

    float Mr[8], iS[8];
#pragma unroll
    for (int r = 0; r < 8; ++r) {
        const int row = hi * 8 + r;
        Mr[r] = finM[row];
        iS[r] = finIS[row];
    }

    // -- pass 2: recompute logits (cheap), write normalized softmax once --
    for (int t = wave; t < OUT_DIM / 16; t += 8) {
        v16h bfrag;
        {
            union { v16h h; f32x4 f[2]; } ub;
            const f32x4* bp = (const f32x4*)(w3p + ((size_t)t * 32 + lane) * 16);
            ub.f[0] = bp[0];
            ub.f[1] = bp[1];
            bfrag = ub.h;
        }
        const int   col  = t * 16 + nlane;
        const float bias = b3[col];
        v8f c;
#pragma unroll
        for (int r = 0; r < 8; ++r) c[r] = bias;

        c = __builtin_amdgcn_wmma_f32_16x16x32_f16(
                false, afrag, false, bfrag, (short)0, c, false, false);

        const int colw = col >> 5;
        const int colb = col & 31;
#pragma unroll
        for (int r = 0; r < 8; ++r) {
            const int row = hi * 8 + r;
            const unsigned int bit = (maskBits[row * 128 + colw] >> colb) & 1u;
            const float f = bit ? c[r] : 0.f;
            out[(size_t)(rowBase + row) * OUT_DIM + col] = __expf(f - Mr[r]) * iS[r];
        }
    }
}

// ---------------------------------------------------------------------------
extern "C" void kernel_launch(void* const* d_in, const int* in_sizes, int n_in,
                              void* d_out, int out_size, void* d_ws, size_t ws_size,
                              hipStream_t stream) {
    const float* x     = (const float*)d_in[0];
    const int*   moves = (const int*)d_in[1];
    const float* W1    = (const float*)d_in[2];
    const float* b1    = (const float*)d_in[3];
    const float* W2    = (const float*)d_in[4];
    const float* b2    = (const float*)d_in[5];
    const float* W3    = (const float*)d_in[6];
    const float* b3    = (const float*)d_in[7];
    float*       out   = (float*)d_out;

    _Float16* hbuf = (_Float16*)d_ws;                                   // 16384*32*2 = 1 MB
    _Float16* w3p  = (_Float16*)((char*)d_ws + (size_t)BATCH * 32 * 2); // 256 KB

    mlp_hidden_kernel<<<BATCH / 256, 256, 0, stream>>>(x, W1, b1, W2, b2, hbuf);
    prep_w3_kernel<<<(OUT_DIM / 16) * 32 / 256, 256, 0, stream>>>(W3, w3p);
    gemm_mask_softmax_kernel<<<BATCH / 16, 256, 0, stream>>>(hbuf, w3p, b3, moves, out);
}
